// PointFeatureNet_47107201303118
// MI455X (gfx1250) — compile-verified, matrix operand order checked by reading, not run on hardware
//
#include <hip/hip_runtime.h>

typedef __attribute__((ext_vector_type(16))) _Float16 v16h;
typedef __attribute__((ext_vector_type(8)))  _Float16 v8h;
typedef __attribute__((ext_vector_type(8)))  float    v8f;

#define K_NBR 64

// ---------------------------------------------------------------- unpack x -> pos(f32), feat(f16)
__global__ void unpack_kernel(const float* __restrict__ x, float* __restrict__ pos,
                              _Float16* __restrict__ feat, int n) {
  int i = blockIdx.x * blockDim.x + threadIdx.x;
  if (i >= n) return;
  pos[i*3+0] = x[i*5+0];
  pos[i*3+1] = x[i*5+1];
  pos[i*3+2] = x[i*5+2];
  feat[i*2+0] = (_Float16)x[i*5+3];
  feat[i*2+1] = (_Float16)x[i*5+4];
}

// ---------------------------------------------------------------- W[cin,cout] f32 -> WT[cout,kpad] f16
__global__ void wconv_kernel(const float* __restrict__ W, _Float16* __restrict__ WT,
                             int cin, int cout, int cinPad) {
  int idx = blockIdx.x * blockDim.x + threadIdx.x;
  if (idx >= cout * cinPad) return;
  int nch = idx / cinPad;
  int k   = idx % cinPad;
  WT[idx] = (k < cin) ? (_Float16)W[(size_t)k * cout + nch] : (_Float16)0.f;
}

// ---------------------------------------------------------------- farthest point sampling (1 block)
__global__ __launch_bounds__(1024)
void fps_kernel(const float* __restrict__ pos, int n, int s,
                int* __restrict__ cidx, float* __restrict__ centers) {
  __shared__ float d[8192];
  __shared__ float sval[32];
  __shared__ int   sidx[32];
  __shared__ float curp[3];
  const int tid = threadIdx.x;
  const int nth = blockDim.x;

  float p0x = pos[0], p0y = pos[1], p0z = pos[2];
  for (int i = tid; i < n; i += nth) {
    float dx = pos[i*3+0] - p0x, dy = pos[i*3+1] - p0y, dz = pos[i*3+2] - p0z;
    d[i] = dx*dx + dy*dy + dz*dz;
  }
  if (tid == 0) {
    cidx[0] = 0;
    centers[0] = p0x; centers[1] = p0y; centers[2] = p0z;
  }
  __syncthreads();

  for (int t = 1; t < s; t++) {
    // block-wide argmax over d[0..n), first-index tie-break
    float bv = -1.f; int bi = 0;
    for (int i = tid; i < n; i += nth) {
      float v = d[i];
      if (v > bv) { bv = v; bi = i; }
    }
    for (int off = 16; off > 0; off >>= 1) {
      float ov = __shfl_down(bv, off, 32);
      int   oi = __shfl_down(bi, off, 32);
      if (ov > bv || (ov == bv && oi < bi)) { bv = ov; bi = oi; }
    }
    int wid = tid >> 5;
    if ((tid & 31) == 0) { sval[wid] = bv; sidx[wid] = bi; }
    __syncthreads();
    if (tid < 32) {
      int nw = nth >> 5;
      bv = (tid < nw) ? sval[tid] : -1.f;
      bi = (tid < nw) ? sidx[tid] : 0;
      for (int off = 16; off > 0; off >>= 1) {
        float ov = __shfl_down(bv, off, 32);
        int   oi = __shfl_down(bi, off, 32);
        if (ov > bv || (ov == bv && oi < bi)) { bv = ov; bi = oi; }
      }
      if (tid == 0) {
        cidx[t] = bi;
        float cx = pos[bi*3+0], cy = pos[bi*3+1], cz = pos[bi*3+2];
        curp[0] = cx; curp[1] = cy; curp[2] = cz;
        centers[t*3+0] = cx; centers[t*3+1] = cy; centers[t*3+2] = cz;
      }
    }
    __syncthreads();
    float cx = curp[0], cy = curp[1], cz = curp[2];
    for (int i = tid; i < n; i += nth) {
      float dx = pos[i*3+0] - cx, dy = pos[i*3+1] - cy, dz = pos[i*3+2] - cz;
      float nd = dx*dx + dy*dy + dz*dz;
      if (nd < d[i]) d[i] = nd;
    }
    __syncthreads();
  }
}

// ---------------------------------------------------------------- ball query: one wave32 per centroid
__global__ void ballquery_kernel(const float* __restrict__ pos, int n,
                                 const float* __restrict__ centers,
                                 const int* __restrict__ cidx, int s, float r2,
                                 int* __restrict__ nbr,
                                 unsigned long long* __restrict__ vmask) {
  int wave = (blockIdx.x * blockDim.x + threadIdx.x) >> 5;
  int lane = threadIdx.x & 31;
  if (wave >= s) return;
  float cx = centers[wave*3+0], cy = centers[wave*3+1], cz = centers[wave*3+2];
  int* out = nbr + (size_t)wave * K_NBR;
  int count = 0;
  for (int j0 = 0; j0 < n && count < K_NBR; j0 += 32) {
    int j = j0 + lane;
    bool ok = false;
    if (j < n) {
      float dx = pos[j*3+0] - cx, dy = pos[j*3+1] - cy, dz = pos[j*3+2] - cz;
      ok = (dx*dx + dy*dy + dz*dz) <= r2;
    }
    unsigned long long m = __ballot(ok);
    int rank = __popcll(m & ((1ull << lane) - 1ull));
    if (ok && (count + rank) < K_NBR) out[count + rank] = j;
    count += __popcll(m);
    if (count > K_NBR) count = K_NBR;
  }
  for (int k = lane; k < K_NBR; k += 32)
    if (k >= count) out[k] = 0;
  if (lane == 0) {
    out[0] = cidx[wave];
    unsigned long long vm = (count >= 64) ? ~0ull : ((1ull << count) - 1ull);
    vmask[wave] = vm | 1ull;
  }
}

// ---------------------------------------------------------------- gather edge features into padded f16 H
__global__ void gather_kernel(const _Float16* __restrict__ feats, int fch,
                              const float* __restrict__ pos,
                              const float* __restrict__ centers,
                              const int* __restrict__ nbr,
                              int c0, int sc, int cinPad,
                              _Float16* __restrict__ H) {
  int e = blockIdx.x * blockDim.x + threadIdx.x;
  if (e >= sc * K_NBR) return;
  int c = c0 + (e >> 6);
  int j = nbr[(size_t)c * K_NBR + (e & 63)];
  _Float16* h = H + (size_t)e * cinPad;
  const _Float16* f = feats + (size_t)j * fch;
  if ((fch & 7) == 0) {
    for (int q = 0; q < fch; q += 8)
      *(v8h*)(h + q) = *(const v8h*)(f + q);
  } else {
    for (int q = 0; q < fch; q++) h[q] = f[q];
  }
  h[fch+0] = (_Float16)(pos[j*3+0] - centers[c*3+0]);
  h[fch+1] = (_Float16)(pos[j*3+1] - centers[c*3+1]);
  h[fch+2] = (_Float16)(pos[j*3+2] - centers[c*3+2]);
  for (int q = fch + 3; q < cinPad; q++) h[q] = (_Float16)0.f;
}

// ---------------------------------------------------------------- Y = relu(X @ W + b), WMMA f16, f32 acc
// X:[E,K] f16, WT:[N,K] f16 (pre-transposed), Y:[E,N] f16.
// E % 64 == 0, K % 32 == 0, N % 32 == 0. One wave per 32x32 output block (2x2 WMMA tiles).
__global__ __launch_bounds__(256)
void gemm_wmma_kernel(const _Float16* __restrict__ X, const _Float16* __restrict__ WT,
                      const float* __restrict__ bias, _Float16* __restrict__ Y,
                      int E, int K, int N) {
  const int lane  = threadIdx.x & 31;
  const int wave  = (blockIdx.x * blockDim.x + threadIdx.x) >> 5;
  const int nsb   = N >> 5;                 // 32-wide super-tiles in N
  const int total = (E >> 5) * nsb;
  if (wave >= total) return;
  const int mt = (wave / nsb) * 2;          // 16-row tile index
  const int nt = (wave % nsb) * 2;          // 16-col tile index

  // A/B 16-bit fragment lane layout (ISA 7.12.2): lane holds row (lane&15),
  // halves 0..7 at K = base0..+7, halves 8..15 at K = base1..+7.
  const int base0 = (lane < 16) ? 0 : 8;
  const int base1 = (lane < 16) ? 16 : 24;
  const _Float16* arow0 = X  + (size_t)((mt + 0) * 16 + (lane & 15)) * K;
  const _Float16* arow1 = X  + (size_t)((mt + 1) * 16 + (lane & 15)) * K;
  const _Float16* brow0 = WT + (size_t)((nt + 0) * 16 + (lane & 15)) * K;
  const _Float16* brow1 = WT + (size_t)((nt + 1) * 16 + (lane & 15)) * K;

  __builtin_prefetch(arow0, 0, 0);          // global_prefetch_b8
  __builtin_prefetch(arow1, 0, 0);

  union Frag { v16h v; v8h h[2]; };
  v8f acc00 = {}, acc01 = {}, acc10 = {}, acc11 = {};
  for (int k0 = 0; k0 < K; k0 += 32) {
    Frag a0, a1, b0, b1;
    a0.h[0] = *(const v8h*)(arow0 + k0 + base0);
    a0.h[1] = *(const v8h*)(arow0 + k0 + base1);
    a1.h[0] = *(const v8h*)(arow1 + k0 + base0);
    a1.h[1] = *(const v8h*)(arow1 + k0 + base1);
    b0.h[0] = *(const v8h*)(brow0 + k0 + base0);
    b0.h[1] = *(const v8h*)(brow0 + k0 + base1);
    b1.h[0] = *(const v8h*)(brow1 + k0 + base0);
    b1.h[1] = *(const v8h*)(brow1 + k0 + base1);
    acc00 = __builtin_amdgcn_wmma_f32_16x16x32_f16(false, a0.v, false, b0.v, (short)0, acc00, false, false);
    acc01 = __builtin_amdgcn_wmma_f32_16x16x32_f16(false, a0.v, false, b1.v, (short)0, acc01, false, false);
    acc10 = __builtin_amdgcn_wmma_f32_16x16x32_f16(false, a1.v, false, b0.v, (short)0, acc10, false, false);
    acc11 = __builtin_amdgcn_wmma_f32_16x16x32_f16(false, a1.v, false, b1.v, (short)0, acc11, false, false);
  }

  // D layout: VGPR v -> m = v + (lane<16 ? 0 : 8), n = lane&15
  const int nloc = lane & 15;
  const int mo   = (lane < 16) ? 0 : 8;
  const int n0   = nt * 16 + nloc;
  const int n1   = n0 + 16;
  const float bn0 = bias[n0];
  const float bn1 = bias[n1];
  const int m0 = (mt + 0) * 16 + mo;
  const int m1 = (mt + 1) * 16 + mo;
#pragma unroll
  for (int v = 0; v < 8; v++) {
    float y00 = acc00[v] + bn0; y00 = (y00 > 0.f) ? y00 : 0.f;
    float y01 = acc01[v] + bn1; y01 = (y01 > 0.f) ? y01 : 0.f;
    float y10 = acc10[v] + bn0; y10 = (y10 > 0.f) ? y10 : 0.f;
    float y11 = acc11[v] + bn1; y11 = (y11 > 0.f) ? y11 : 0.f;
    Y[(size_t)(m0 + v) * N + n0] = (_Float16)y00;
    Y[(size_t)(m0 + v) * N + n1] = (_Float16)y01;
    Y[(size_t)(m1 + v) * N + n0] = (_Float16)y10;
    Y[(size_t)(m1 + v) * N + n1] = (_Float16)y11;
  }
}

// ---------------------------------------------------------------- masked max over K=64 neighbors, 8 ch/thread
__global__ void maxpool_kernel(const _Float16* __restrict__ Y,
                               const unsigned long long* __restrict__ vmask,
                               int c0, int sc, int N,
                               _Float16* __restrict__ out16,
                               float* __restrict__ out32) {
  int idx = blockIdx.x * blockDim.x + threadIdx.x;
  int nv = N >> 3;
  if (idx >= sc * nv) return;
  int ci  = idx / nv;
  int ch8 = (idx % nv) * 8;
  unsigned long long vm = vmask[c0 + ci];
  const _Float16* y = Y + (size_t)ci * K_NBR * N + ch8;
  float m[8];
#pragma unroll
  for (int j = 0; j < 8; j++) m[j] = -1e30f;
  for (int k = 0; k < K_NBR; k++) {
    if (vm & (1ull << k)) {
      v8h v = *(const v8h*)(y + (size_t)k * N);
#pragma unroll
      for (int j = 0; j < 8; j++) {
        float f = (float)v[j];
        if (f > m[j]) m[j] = f;
      }
    }
  }
  if (out16) {
    v8h o;
#pragma unroll
    for (int j = 0; j < 8; j++) o[j] = (_Float16)m[j];
    *(v8h*)(out16 + (size_t)(c0 + ci) * N + ch8) = o;
  }
  if (out32) {
    float* o = out32 + (size_t)(c0 + ci) * N + ch8;
#pragma unroll
    for (int j = 0; j < 8; j++) o[j] = m[j];
  }
}

// ================================================================ host orchestration
static inline void launch_gemm(const _Float16* X, const _Float16* WT, const float* bias,
                               _Float16* Y, int E, int K, int N, hipStream_t stream) {
  int totalWaves = (E / 32) * (N / 32);
  int blocks = (totalWaves + 7) / 8;
  gemm_wmma_kernel<<<blocks, 256, 0, stream>>>(X, WT, bias, Y, E, K, N);
}

extern "C" void kernel_launch(void* const* d_in, const int* in_sizes, int n_in,
                              void* d_out, int out_size, void* d_ws, size_t ws_size,
                              hipStream_t stream) {
  (void)in_sizes; (void)n_in; (void)out_size; (void)ws_size;
  const int Bb = 2, Npts = 8192;
  static const int   nl[4]   = {8192, 4096, 1024, 256};
  static const int   sl[4]   = {4096, 1024, 256, 64};
  static const float rl[4]   = {0.5f, 1.0f, 2.0f, 4.0f};
  static const int   fch[4]  = {2, 64, 128, 256};
  static const int   dims[4][4] = {{5,32,32,64},{67,64,64,128},{131,128,128,256},{259,256,256,512}};
  static const int   kpad[4] = {32, 96, 160, 288};
  const int CH = 512; // centroid chunk

  char* ws = (char*)d_ws;
  size_t cur = 0;
  auto alloc = [&](size_t bytes) -> char* {
    char* p = ws + cur;
    cur += (bytes + 255) & ~(size_t)255;
    return p;
  };

  _Float16* wt[4][3];
  for (int L = 0; L < 4; L++)
    for (int j = 0; j < 3; j++) {
      int kin  = (j == 0) ? kpad[L] : dims[L][j];
      int kout = dims[L][j+1];
      wt[L][j] = (_Float16*)alloc((size_t)kin * kout * sizeof(_Float16));
    }
  float*    posA = (float*)alloc((size_t)8192 * 3 * sizeof(float));
  float*    posB = (float*)alloc((size_t)8192 * 3 * sizeof(float));
  _Float16* fA   = (_Float16*)alloc((size_t)4096 * 64 * sizeof(_Float16));
  _Float16* fB   = (_Float16*)alloc((size_t)4096 * 64 * sizeof(_Float16));
  int*      cidx = (int*)alloc((size_t)4096 * sizeof(int));
  int*      nbr  = (int*)alloc((size_t)4096 * K_NBR * sizeof(int));
  unsigned long long* vmask = (unsigned long long*)alloc((size_t)4096 * 8);
  _Float16* Hbuf = (_Float16*)alloc((size_t)32768 * 96  * sizeof(_Float16)); // max chunk H
  _Float16* Ya   = (_Float16*)alloc((size_t)32768 * 128 * sizeof(_Float16)); // max chunk Y
  _Float16* Yb   = (_Float16*)alloc((size_t)32768 * 128 * sizeof(_Float16));

  // convert weights (deterministic, every call)
  for (int L = 0; L < 4; L++)
    for (int j = 0; j < 3; j++) {
      const float* W = (const float*)d_in[1 + (3*L + j) * 2];
      int cin = dims[L][j], cout = dims[L][j+1];
      int kin = (j == 0) ? kpad[L] : dims[L][j];
      int tot = cout * kin;
      wconv_kernel<<<(tot + 255) / 256, 256, 0, stream>>>(W, wt[L][j], cin, cout, kin);
    }

  for (int b = 0; b < Bb; b++) {
    const float* x = (const float*)d_in[0] + (size_t)b * Npts * 5;
    unpack_kernel<<<(Npts + 255) / 256, 256, 0, stream>>>(x, posA, fA, Npts);

    float*    posCur = posA; float*    posNxt = posB;
    _Float16* fCur   = fA;   _Float16* fNxt   = fB;

    for (int L = 0; L < 4; L++) {
      int n = nl[L], s = sl[L];
      fps_kernel<<<1, 1024, 0, stream>>>(posCur, n, s, cidx, posNxt);
      ballquery_kernel<<<(s * 32 + 255) / 256, 256, 0, stream>>>(
          posCur, n, posNxt, cidx, s, rl[L] * rl[L], nbr, vmask);

      int d1 = dims[L][1], d2 = dims[L][2], d3 = dims[L][3];
      const float* b0 = (const float*)d_in[2 + (3*L + 0) * 2];
      const float* b1 = (const float*)d_in[2 + (3*L + 1) * 2];
      const float* b2 = (const float*)d_in[2 + (3*L + 2) * 2];

      for (int c0 = 0; c0 < s; c0 += CH) {
        int sc = (s - c0 < CH) ? (s - c0) : CH;
        int E  = sc * K_NBR;
        gather_kernel<<<(E + 255) / 256, 256, 0, stream>>>(
            fCur, fch[L], posCur, posNxt, nbr, c0, sc, kpad[L], Hbuf);
        launch_gemm(Hbuf, wt[L][0], b0, Ya, E, kpad[L], d1, stream);
        launch_gemm(Ya,   wt[L][1], b1, Yb, E, d1,      d2, stream);
        launch_gemm(Yb,   wt[L][2], b2, Ya, E, d2,      d3, stream);
        _Float16* o16 = (L < 3) ? fNxt : (_Float16*)nullptr;
        float*    o32 = (L == 3) ? ((float*)d_out + (size_t)b * 64 * 512) : (float*)nullptr;
        maxpool_kernel<<<((sc * (d3 >> 3)) + 255) / 256, 256, 0, stream>>>(
            Ya, vmask, c0, sc, d3, o16, o32);
      }
      float* tp = posCur; posCur = posNxt; posNxt = tp;
      _Float16* tf = fCur; fCur = fNxt; fNxt = tf;
    }
  }
}
